// TurboQuantMSE_2250562863575
// MI455X (gfx1250) — matrix operand order repaired and checked by reading, not compiled, not run on hardware
//
#include <hip/hip_runtime.h>

// ---- CDNA5 WMMA types ------------------------------------------------------
typedef __attribute__((ext_vector_type(16))) _Float16 half16;
typedef __attribute__((ext_vector_type(8)))  _Float16 half8;
typedef __attribute__((ext_vector_type(8)))  float    float8;
typedef __attribute__((ext_vector_type(4)))  float    float4v;
typedef __attribute__((ext_vector_type(4)))  int      int4v;

union FragU { half16 h; int4v i[2]; };

#define WAVES_PER_BLOCK 4
#define ROWS_PER_WAVE   16
#define DHEAD           128
#define LDS_STRIDE      136   // 128 + 8 halves pad -> balanced LDS banks

// ---- Prep: convert Pi (f32, row-major) into two f16 A-operand matrices -----
// GEMM1 (transposed): rotated^T = Pi   @ Xn^T -> A rows = Pi[n][k]   = PiA row-major
// GEMM2 (transposed): recon^T   = Pi^T @ C^T  -> A rows = Pi[k][d]^T = PiB row-major
__global__ void tq_prep_kernel(const float* __restrict__ Pi,
                               _Float16* __restrict__ PiA,
                               _Float16* __restrict__ PiB) {
  int t = blockIdx.x * blockDim.x + threadIdx.x;
  for (int i = t; i < DHEAD * DHEAD; i += gridDim.x * blockDim.x) {
    int n = i >> 7;
    int k = i & (DHEAD - 1);
    PiA[i] = (_Float16)Pi[i];               // row n: Pi[n][k], K contiguous
    PiB[i] = (_Float16)Pi[k * DHEAD + n];   // row d: Pi[k][d], K contiguous
  }
}

// ---- Main: norm -> rotate (WMMA) -> quantize -> unrotate (WMMA) -> scale ---
__launch_bounds__(WAVES_PER_BLOCK * 32)
__global__ void tq_main_kernel(const float* __restrict__ x,
                               const _Float16* __restrict__ PiA,
                               const _Float16* __restrict__ PiB,
                               const float* __restrict__ centroids,
                               const float* __restrict__ boundaries,
                               float* __restrict__ out) {
  __shared__ __attribute__((aligned(16))) _Float16 sXH[WAVES_PER_BLOCK][ROWS_PER_WAVE][LDS_STRIDE];
  __shared__ __attribute__((aligned(16))) _Float16 sCH[WAVES_PER_BLOCK][ROWS_PER_WAVE][LDS_STRIDE];

  const int  w    = threadIdx.x >> 5;
  const int  lane = threadIdx.x & 31;
  const int  m    = lane & 15;      // this lane's row within the 16-row tile
  const int  hi   = lane >> 4;      // half-wave selector
  const long r0   = (long)blockIdx.x * (WAVES_PER_BLOCK * ROWS_PER_WAVE)
                    + (long)w * ROWS_PER_WAVE;
  const long row  = r0 + m;

  // Symmetric quantizer tables (cen[15-j] = -cen[j], bnd[8] = 0):
  // compare |r| against upper 7 boundaries, select among upper 8 centroids,
  // then restore the sign. Uniform addresses -> scalar regs.
  float bndA[7], cenA[8];
#pragma unroll
  for (int j = 0; j < 7; ++j) bndA[j] = boundaries[9 + j];  // bnd[9..15]
#pragma unroll
  for (int j = 0; j < 8; ++j) cenA[j] = centroids[8 + j];   // cen[8..15]

  // ---------------- Phase 1: load, norm, normalize -> f16 LDS ----------------
  // x is streamed exactly once -> non-temporal loads (TH=NT) so the 256 MB
  // stream does not evict the hot Pi operand tiles from L1/L2.
  float nf;  // fp16-rounded norm (this lane's own row)
  {
    const float4v* xp = (const float4v*)(x + row * DHEAD + hi * 64);
    float4v xf[16];
    float ss = 0.0f;
#pragma unroll
    for (int i = 0; i < 16; ++i) {
      float4v v = __builtin_nontemporal_load(xp + i);
      xf[i] = v;
      ss += v[0] * v[0] + v[1] * v[1] + v[2] * v[2] + v[3] * v[3];
    }
    ss += __shfl_xor(ss, 16, 32);          // combine the two half-rows
    float norm = sqrtf(ss);
    float inv  = 1.0f / (norm + 1e-8f);
    nf = (float)(_Float16)norm;            // norms stored as fp16 in reference

#pragma unroll
    for (int c = 0; c < 8; ++c) {
      float4v a = xf[2 * c], b = xf[2 * c + 1];
      half8 h;
      h[0] = (_Float16)(a[0] * inv); h[1] = (_Float16)(a[1] * inv);
      h[2] = (_Float16)(a[2] * inv); h[3] = (_Float16)(a[3] * inv);
      h[4] = (_Float16)(b[0] * inv); h[5] = (_Float16)(b[1] * inv);
      h[6] = (_Float16)(b[2] * inv); h[7] = (_Float16)(b[3] * inv);
      *(half8*)&sXH[w][m][hi * 64 + c * 8] = h;
    }
  }
  // Wave-private LDS slice: wave-level ordering is enough (LDS ops from one
  // wave are in-order; the asm is a compiler memory barrier + DS drain).
  asm volatile("s_wait_dscnt 0x0" ::: "memory");
  __builtin_amdgcn_wave_barrier();

  // ------- Phase 2: rotated^T = Pi @ Xn^T  (A=PiA rows, B=sXH rows) ----------
  // D layout: lane holds col m (its own row), rows 8*hi+e -> n = 16*nb+8*hi+e,
  // i.e. 8 *contiguous* rotated coords -> one packed b128 store to sCH.
#pragma unroll 1
  for (int nb = 0; nb < 8; ++nb) {
    float8 acc = {0.f, 0.f, 0.f, 0.f, 0.f, 0.f, 0.f, 0.f};
#pragma unroll
    for (int kb = 0; kb < 4; ++kb) {
      FragU a, b;
      // A frag (16x32): lane row = 16*nb + m; VGPR0-3 <- K=32kb+8hi..+7, VGPR4-7 <- +16
      const _Float16* arow = PiA + (long)(nb * 16 + m) * DHEAD + kb * 32 + hi * 8;
      a.i[0] = *(const int4v*)(arow);
      a.i[1] = *(const int4v*)(arow + 16);
      // B frag (32x16): lane col = m; VGPR0-3 <- K=32kb+8hi..+7, VGPR4-7 <- +16
      const _Float16* brow = &sXH[w][m][kb * 32 + hi * 8];
      b.i[0] = *(const int4v*)(brow);
      b.i[1] = *(const int4v*)(brow + 16);
      acc = __builtin_amdgcn_wmma_f32_16x16x32_f16(
          false, a.h, false, b.h, (short)0, acc, false, false);
    }
    // symmetric nearest-centroid quantize (== searchsorted midpoints, left)
    half8 q;
#pragma unroll
    for (int e = 0; e < 8; ++e) {
      float r = acc[e];
      float ar = fabsf(r);                 // folds into v_cmp |.| modifier
      float c = cenA[0];
#pragma unroll
      for (int j = 0; j < 7; ++j) c = (ar > bndA[j]) ? cenA[j + 1] : c;
      q[e] = (_Float16)__builtin_copysignf(c, r);
    }
    *(half8*)&sCH[w][m][nb * 16 + hi * 8] = q;   // one b128 store
  }
  asm volatile("s_wait_dscnt 0x0" ::: "memory");
  __builtin_amdgcn_wave_barrier();

  // ------- Phase 3: recon^T = Pi^T @ C^T  (A=PiB rows, B=sCH rows) -----------
  // D layout: lane holds out row m, cols 16*db+8*hi+e -> 8 contiguous floats.
  // out is written once and never re-read -> non-temporal stores.
#pragma unroll 1
  for (int db = 0; db < 8; ++db) {
    float8 acc = {0.f, 0.f, 0.f, 0.f, 0.f, 0.f, 0.f, 0.f};
#pragma unroll
    for (int kb = 0; kb < 4; ++kb) {
      FragU a, b;
      const _Float16* arow = PiB + (long)(db * 16 + m) * DHEAD + kb * 32 + hi * 8;
      a.i[0] = *(const int4v*)(arow);
      a.i[1] = *(const int4v*)(arow + 16);
      const _Float16* brow = &sCH[w][m][kb * 32 + hi * 8];
      b.i[0] = *(const int4v*)(brow);
      b.i[1] = *(const int4v*)(brow + 16);
      acc = __builtin_amdgcn_wmma_f32_16x16x32_f16(
          false, a.h, false, b.h, (short)0, acc, false, false);
    }
    float4v o0, o1;
    o0[0] = acc[0] * nf; o0[1] = acc[1] * nf; o0[2] = acc[2] * nf; o0[3] = acc[3] * nf;
    o1[0] = acc[4] * nf; o1[1] = acc[5] * nf; o1[2] = acc[6] * nf; o1[3] = acc[7] * nf;
    float4v* op = (float4v*)(out + row * DHEAD + db * 16 + hi * 8);
    __builtin_nontemporal_store(o0, op);
    __builtin_nontemporal_store(o1, op + 1);
  }
}

// ---- Harness entry ---------------------------------------------------------
extern "C" void kernel_launch(void* const* d_in, const int* in_sizes, int n_in,
                              void* d_out, int out_size, void* d_ws, size_t ws_size,
                              hipStream_t stream) {
  const float* x          = (const float*)d_in[0];
  const float* Pi         = (const float*)d_in[1];
  const float* centroids  = (const float*)d_in[2];
  const float* boundaries = (const float*)d_in[3];
  float*       out        = (float*)d_out;

  _Float16* PiA = (_Float16*)d_ws;                   // 128*128 f16 = 32 KB
  _Float16* PiB = PiA + DHEAD * DHEAD;               // 32 KB

  tq_prep_kernel<<<32, 256, 0, stream>>>(Pi, PiA, PiB);

  long nrows  = (long)in_sizes[0] / DHEAD;           // 524288
  int  blocks = (int)(nrows / (WAVES_PER_BLOCK * ROWS_PER_WAVE));  // 8192
  tq_main_kernel<<<blocks, WAVES_PER_BLOCK * 32, 0, stream>>>(
      x, PiA, PiB, centroids, boundaries, out);

  (void)n_in; (void)out_size; (void)ws_size;
}
